// RegionConsistencyModule_76184129896954
// MI455X (gfx1250) — matrix operand order repaired and checked by reading, not compiled, not run on hardware
//
#include <hip/hip_runtime.h>
#include <hip/hip_bf16.h>
#include <math.h>

// ---------------------------------------------------------------------------
// RegionConsistencyModule for MI455X (gfx1250, wave32, WMMA)
// ---------------------------------------------------------------------------

typedef __attribute__((ext_vector_type(16))) __bf16 bf16x16;
typedef __attribute__((ext_vector_type(8)))  float  f32x8;

#define C_CH   256
#define HWN    4096          // H*W = 64*64
#define HDIM   64
#define BATCH  8
#define BN_EPS 1e-5f

// GEMM tiling: WG = 256 thr = 8 wave32s; WG tile M=64 x N=256; K-step 32.
// Wave tile 32x64: 2 A-fragments x 4 B-fragments -> 8 WMMAs per K-step.
#define TM 64
#define TN 256
#define TK 32
#define KPAD 8                       // bf16 elems of row padding
#define KROW (TK + KPAD)             // 40 bf16 = 80 B (16B aligned stride)

__device__ __forceinline__ unsigned pack_bf16x2(float a, float b)
{
    __bf16 ha = (__bf16)a;
    __bf16 hb = (__bf16)b;
    unsigned short ua = __builtin_bit_cast(unsigned short, ha);
    unsigned short ub = __builtin_bit_cast(unsigned short, hb);
    return (unsigned)ua | ((unsigned)ub << 16);
}

// ---------------------------------------------------------------------------
// Kernel 1/3: Y[b][co][n] = relu(BN(sum_ci W[co][ci] * Xcat[b][ci][n] + bias))
// Xcat channel ci < 256 from X0, ci >= 256 from X1 (concat(x, agg)).
// bf16 WMMA 16x16x32, f32 accumulation. LDS is laid out fragment-major so
// every WMMA operand fragment is exactly two ds_load_b128s.
// ---------------------------------------------------------------------------
__global__ __launch_bounds__(256)
void gemm_bn_relu_kernel(const float* __restrict__ X0,
                         const float* __restrict__ X1,
                         const float* __restrict__ W,
                         const float* __restrict__ bias,
                         const float* __restrict__ gamma,
                         const float* __restrict__ beta,
                         const float* __restrict__ mean,
                         const float* __restrict__ var,
                         float* __restrict__ Y,
                         int Cin)
{
    // X tile stored TRANSPOSED: row n holds K-major run of 32 bf16.
    __shared__ alignas(16) __bf16 sXT[TN][KROW];   // 256*40*2 = 20 KB
    // W tile row-major: row m holds K-major run of 32 bf16.
    __shared__ alignas(16) __bf16 sW[TM][KROW];    //  64*40*2 =  5 KB

    const int tid  = threadIdx.x;
    const int lane = tid & 31;
    const int wave = tid >> 5;
    const int wm   = wave >> 2;      // 0..1 : 32-row slab
    const int wn   = wave & 3;       // 0..3 : 64-col slab

    const int n0 = blockIdx.x * TN;  // pixel base within batch
    const int m0 = blockIdx.y * TM;  // output-channel base
    const int b  = blockIdx.z;

    f32x8 acc[2][4];
    #pragma unroll
    for (int a = 0; a < 2; ++a)
        #pragma unroll
        for (int t = 0; t < 4; ++t)
            #pragma unroll
            for (int r = 0; r < 8; ++r) acc[a][t][r] = 0.0f;

    // lane mappings, contiguous k = 16*(lane/16) + s for both A 16x32 and B 32x16
    const int mlane = lane & 15;
    const int kbA   = (lane < 16) ? 0 : 16;
    const int kbB   = (lane < 16) ? 0 : 16;
    const int nlane = lane & 15;

    unsigned* sXT32 = (unsigned*)&sXT[0][0];       // row stride KROW/2 uints
    const int wrow = tid >> 2;                     // W-stage: row per 4 thr
    const int wk0  = (tid & 3) * 8;                // 8 k-values each

    for (int kc = 0; kc < Cin; kc += TK) {
        // ---- select concat source once per K tile (TK divides the split)
        const float* src = (kc < C_CH)
            ? (X0 + ((size_t)(b * C_CH + kc)) * HWN)
            : (X1 + ((size_t)(b * C_CH + (kc - C_CH))) * HWN);

        // ---- stage X tile transposed: thread owns column n = tid
        #pragma unroll
        for (int k = 0; k < TK; k += 2) {
            const float f0 = src[(size_t)k * HWN + n0 + tid];
            const float f1 = src[(size_t)(k + 1) * HWN + n0 + tid];
            sXT32[tid * (KROW / 2) + (k >> 1)] = pack_bf16x2(f0, f1);
        }
        // prefetch next K tile (global_prefetch_b8)
        if (kc + TK < Cin) {
            const float* nsrc = (kc + TK < C_CH)
                ? (X0 + ((size_t)(b * C_CH + kc + TK)) * HWN)
                : (X1 + ((size_t)(b * C_CH + (kc + TK - C_CH))) * HWN);
            __builtin_prefetch(nsrc + n0 + tid, 0, 1);
        }

        // ---- stage W tile (64 x 32): thread -> (row, 8 k-values)
        {
            const float* wsrc = W + (size_t)(m0 + wrow) * Cin + (kc + wk0);
            unsigned* wdst = (unsigned*)&sW[wrow][wk0];
            #pragma unroll
            for (int i = 0; i < 4; ++i)
                wdst[i] = pack_bf16x2(wsrc[2 * i], wsrc[2 * i + 1]);
        }
        __syncthreads();

        // ---- B fragments (shared by both A rows): 2 x ds_load_b128 each
        union Frag { uint4 q[2]; bf16x16 v; };
        Frag fb[4];
        #pragma unroll
        for (int t = 0; t < 4; ++t) {
            const int n = wn * 64 + t * 16 + nlane;
            fb[t].q[0] = *(const uint4*)&sXT[n][kbB];
            fb[t].q[1] = *(const uint4*)&sXT[n][kbB + 8];
        }

        // ---- A fragments + 8 WMMAs
        #pragma unroll
        for (int a = 0; a < 2; ++a) {
            const int row = wm * 32 + a * 16 + mlane;
            Frag fa;
            fa.q[0] = *(const uint4*)&sW[row][kbA];
            fa.q[1] = *(const uint4*)&sW[row][kbA + 8];
            #pragma unroll
            for (int t = 0; t < 4; ++t) {
                acc[a][t] = __builtin_amdgcn_wmma_f32_16x16x32_bf16(
                    false, fa.v, false, fb[t].v, (short)0, acc[a][t],
                    false, false);
            }
        }
        __syncthreads();
    }

    // ---- fused epilogue: conv-bias folded into BN affine, then ReLU
    const int mh = (lane < 16) ? 0 : 8;    // C/D layout: VGPR r -> M = r + mh
    #pragma unroll
    for (int a = 0; a < 2; ++a) {
        #pragma unroll
        for (int r = 0; r < 8; ++r) {
            const int co  = m0 + wm * 32 + a * 16 + r + mh;
            const float inv = gamma[co] * rsqrtf(var[co] + BN_EPS);
            const float sh  = beta[co] + (bias[co] - mean[co]) * inv;
            #pragma unroll
            for (int t = 0; t < 4; ++t) {
                const int n = n0 + wn * 64 + t * 16 + nlane;
                const float v = acc[a][t][r] * inv + sh;
                Y[((size_t)(b * C_CH + co)) * HWN + n] = fmaxf(v, 0.0f);
            }
        }
    }
}

// ---------------------------------------------------------------------------
// Kernel 2: cosine similarity over 3x3 window + softmax + aggregation of x.
// Thread per pixel; channel loop keeps 19 accumulators in registers.
// ---------------------------------------------------------------------------
__global__ __launch_bounds__(256)
void sim_agg_kernel(const float* __restrict__ fe,
                    const float* __restrict__ x,
                    float* __restrict__ agg)
{
    const int p  = blockIdx.x * blockDim.x + threadIdx.x;   // 0..32767
    const int b  = p >> 12;
    const int hw = p & (HWN - 1);
    const int h  = hw >> 6;
    const int w  = hw & 63;

    int  off[9];
    bool valid[9];
    #pragma unroll
    for (int k = 0; k < 9; ++k) {
        const int di = k / 3 - 1, dj = k % 3 - 1;
        const int hh = h + di, ww = w + dj;
        valid[k] = (hh >= 0) & (hh < HDIM) & (ww >= 0) & (ww < HDIM);
        off[k]   = di * HDIM + dj;
    }

    const float* feb = fe + (size_t)b * C_CH * HWN;
    float dot[9], npw[9], nc = 0.0f;
    #pragma unroll
    for (int k = 0; k < 9; ++k) { dot[k] = 0.0f; npw[k] = 0.0f; }

    for (int c = 0; c < C_CH; ++c) {
        const float* fc = feb + (size_t)c * HWN + hw;
        const float cf = fc[0];
        nc += cf * cf;
        #pragma unroll
        for (int k = 0; k < 9; ++k) {
            const float pv = valid[k] ? fc[off[k]] : 0.0f;
            dot[k] += cf * pv;
            npw[k] += pv * pv;
        }
    }

    const float ncs = sqrtf(nc);
    float sim[9], mx = -1e30f;
    #pragma unroll
    for (int k = 0; k < 9; ++k) {
        sim[k] = dot[k] / (ncs * sqrtf(npw[k]) + 1e-7f);
        mx = fmaxf(mx, sim[k]);
    }
    float se = 0.0f;
    #pragma unroll
    for (int k = 0; k < 9; ++k) { sim[k] = expf(sim[k] - mx); se += sim[k]; }
    const float sinv = 1.0f / se;
    #pragma unroll
    for (int k = 0; k < 9; ++k) sim[k] *= sinv;

    const float* xb = x   + (size_t)b * C_CH * HWN;
    float*       ab = agg + (size_t)b * C_CH * HWN;
    for (int c = 0; c < C_CH; ++c) {
        const float* xc = xb + (size_t)c * HWN + hw;
        float a = 0.0f;
        #pragma unroll
        for (int k = 0; k < 9; ++k)
            a += sim[k] * (valid[k] ? xc[off[k]] : 0.0f);
        ab[(size_t)c * HWN + hw] = a;
    }
}

// ---------------------------------------------------------------------------
// Kernel 4: mean over H,W for concat(x, enhanced) -> pooled[b][0..511]
// ---------------------------------------------------------------------------
__global__ __launch_bounds__(256)
void pool_kernel(const float* __restrict__ x,
                 const float* __restrict__ enh,
                 float* __restrict__ pooled)
{
    __shared__ float red[256];
    const int cc = blockIdx.x & 511;
    const int b  = blockIdx.x >> 9;
    const float* src = (cc < C_CH)
        ? (x   + ((size_t)(b * C_CH + cc)) * HWN)
        : (enh + ((size_t)(b * C_CH + (cc - C_CH))) * HWN);

    float s = 0.0f;
    for (int i = threadIdx.x; i < HWN; i += 256) s += src[i];
    red[threadIdx.x] = s;
    __syncthreads();
    #pragma unroll
    for (int st = 128; st > 0; st >>= 1) {
        if ((int)threadIdx.x < st) red[threadIdx.x] += red[threadIdx.x + st];
        __syncthreads();
    }
    if (threadIdx.x == 0) pooled[b * 512 + cc] = red[0] * (1.0f / HWN);
}

// ---------------------------------------------------------------------------
// Kernel 5: SE gate MLP per batch: relu(W1*pooled+b1) -> sigmoid(W2*h+b2)
// ---------------------------------------------------------------------------
__global__ __launch_bounds__(256)
void gate_kernel(const float* __restrict__ pooled,
                 const float* __restrict__ w1, const float* __restrict__ b1,
                 const float* __restrict__ w2, const float* __restrict__ b2,
                 float* __restrict__ wgt)
{
    __shared__ float sp[512];
    __shared__ float sh[64];
    const int b   = blockIdx.x;
    const int tid = threadIdx.x;

    sp[tid]       = pooled[b * 512 + tid];
    sp[tid + 256] = pooled[b * 512 + tid + 256];
    __syncthreads();

    if (tid < 64) {
        float a = b1[tid];
        for (int c = 0; c < 512; ++c) a += w1[tid * 512 + c] * sp[c];
        sh[tid] = fmaxf(a, 0.0f);
    }
    __syncthreads();

    float a = b2[tid];
    #pragma unroll
    for (int j = 0; j < 64; ++j) a += w2[tid * 64 + j] * sh[j];
    wgt[b * C_CH + tid] = 1.0f / (1.0f + expf(-a));
}

// ---------------------------------------------------------------------------
// Kernel 6: out = x + wgt[b][c] * enhanced   (float4 vectorized)
// ---------------------------------------------------------------------------
__global__ __launch_bounds__(256)
void final_kernel(const float* __restrict__ x,
                  const float* __restrict__ enh,
                  const float* __restrict__ wgt,
                  float* __restrict__ out)
{
    const int i = blockIdx.x * blockDim.x + threadIdx.x;  // float4 index
    const int nvec = (BATCH * C_CH * HWN) / 4;            // 2,097,152
    if (i >= nvec) return;
    const int c = (i >> 10) & 255;   // HWN/4 = 1024 vecs per channel
    const int b = i >> 18;           // 256*1024 vecs per batch
    const float g = wgt[b * C_CH + c];

    const float4 xx = ((const float4*)x)[i];
    const float4 ee = ((const float4*)enh)[i];
    float4 o;
    o.x = xx.x + g * ee.x;
    o.y = xx.y + g * ee.y;
    o.z = xx.z + g * ee.z;
    o.w = xx.w + g * ee.w;
    ((float4*)out)[i] = o;
}

// ---------------------------------------------------------------------------
extern "C" void kernel_launch(void* const* d_in, const int* in_sizes, int n_in,
                              void* d_out, int out_size, void* d_ws, size_t ws_size,
                              hipStream_t stream)
{
    (void)in_sizes; (void)n_in; (void)out_size; (void)ws_size;

    const float* x       = (const float*)d_in[0];
    const float* w_embed = (const float*)d_in[1];
    const float* b_embed = (const float*)d_in[2];
    const float* bn1g    = (const float*)d_in[3];
    const float* bn1b    = (const float*)d_in[4];
    const float* bn1m    = (const float*)d_in[5];
    const float* bn1v    = (const float*)d_in[6];
    const float* w_enh   = (const float*)d_in[7];
    const float* b_enh   = (const float*)d_in[8];
    const float* bn2g    = (const float*)d_in[9];
    const float* bn2b    = (const float*)d_in[10];
    const float* bn2m    = (const float*)d_in[11];
    const float* bn2v    = (const float*)d_in[12];
    const float* w_g1    = (const float*)d_in[13];
    const float* b_g1    = (const float*)d_in[14];
    const float* w_g2    = (const float*)d_in[15];
    const float* b_g2    = (const float*)d_in[16];

    const size_t tensor_elems = (size_t)BATCH * C_CH * HWN;  // 8,388,608

    // workspace layout (fe buffer is reused for 'enhanced' after sim_agg)
    float* fe     = (float*)d_ws;
    float* agg    = fe + tensor_elems;
    float* pooled = agg + tensor_elems;
    float* wgt    = pooled + BATCH * 512;
    float* out    = (float*)d_out;

    const dim3 gG(HWN / TN, C_CH / TM, BATCH);   // 16 x 4 x 8 workgroups

    // 1) fe = relu(BN1(conv1x1(x; w_embed) + b_embed))
    gemm_bn_relu_kernel<<<gG, 256, 0, stream>>>(
        x, x, w_embed, b_embed, bn1g, bn1b, bn1m, bn1v, fe, C_CH);

    // 2) cosine-similarity softmax over 3x3 window, aggregate x -> agg
    sim_agg_kernel<<<(BATCH * HWN) / 256, 256, 0, stream>>>(fe, x, agg);

    // 3) enhanced = relu(BN2(conv1x1(concat(x, agg); w_enh) + b_enh))
    //    (writes over fe buffer; fe is dead after step 2)
    gemm_bn_relu_kernel<<<gG, 256, 0, stream>>>(
        x, agg, w_enh, b_enh, bn2g, bn2b, bn2m, bn2v, fe, 2 * C_CH);

    // 4) pooled[b][0..511] = mean over H,W of concat(x, enhanced)
    pool_kernel<<<BATCH * 512, 256, 0, stream>>>(x, fe, pooled);

    // 5) SE gate: wgt = sigmoid(W2 * relu(W1 * pooled + b1) + b2)
    gate_kernel<<<BATCH, 256, 0, stream>>>(pooled, w_g1, b_g1, w_g2, b_g2, wgt);

    // 6) out = x + wgt * enhanced
    final_kernel<<<((BATCH * C_CH * HWN) / 4 + 255) / 256, 256, 0, stream>>>(
        x, fe, wgt, out);
}
